// StatePredictor_28183575396754
// MI455X (gfx1250) — matrix operand order repaired and checked by reading, not compiled, not run on hardware
//
#include <hip/hip_runtime.h>

// ---------------- CDNA5 WMMA / TDM types ----------------
typedef __attribute__((ext_vector_type(16))) __bf16 v16bf;
typedef __attribute__((ext_vector_type(8)))  float  v8f;
typedef unsigned int u32x4 __attribute__((ext_vector_type(4)));
typedef int          i32x4 __attribute__((ext_vector_type(4)));
typedef int          i32x8 __attribute__((ext_vector_type(8)));

#define NT   160      // threads per block (5 wave32 waves)
#define TOK  80       // tokens per workgroup (4 batches * 20)
#define NB   4        // batches per workgroup
#define SEQ  20       // tokens per batch

// ---------------- pre-converted weight images in d_ws ----------------
// Execution order: m1_w0, m1_w1, m2_w0, m2_w1, at_w0[:100], at_w1, at_w2,
//                  hp_w0[:50], hp_w1, hp_w2   — each [npad][kpad] bf16, [n][k].
constexpr int IMG_OFF[10] = {0, 5120, 25600, 41984, 50176, 66560, 82944, 84992, 93184, 109568};
constexpr int IMG_SZ [10] = {5120, 20480, 16384, 8192, 16384, 16384, 2048, 8192, 16384, 2048};
constexpr int WIMG_TOTAL  = 111616;                  // bf16 elems
constexpr size_t WS_WIMG  = 0;                       // bytes
constexpr size_t WS_XBF   = 223232;                  // 655360*16 bf16 = 20971520 B
constexpr size_t WS_MASK  = 223232 + 20971520;       // 655360 f32

// biases staged to LDS zero-padded to each layer's padded output width
constexpr int BO[11] = {0, 160, 288, 416, 480, 608, 736, 752, 880, 1008, 1024};
constexpr int BFO[10] = {150, 100, 100, 50, 100, 100, 1, 100, 100, 7};

// ---------------- LDS layout (bytes) ----------------
constexpr int OFF_WBUF0 = 0;                          // 40960 B
constexpr int OFF_WBUF1 = 40960;                      // 40960 B
constexpr int OFF_X     = 81920;                      // [80][32]  bf16 = 5120
constexpr int OFF_T1    = 87040;                      // [80][160] bf16 = 25600
constexpr int OFF_H1    = 112640;                     // [80][128] bf16 = 20480
constexpr int OFF_H2    = 133120;                     // [80][64]  bf16 = 10240
constexpr int OFF_SC    = 143360;                     // [80][16]  bf16 = 2560
constexpr int OFF_MASK  = 145920;                     // [80] f32
constexpr int OFF_MSUM  = 146240;                     // [16] f32
constexpr int OFF_G     = 146304;                     // [4][100] f32
constexpr int OFF_GB    = 147904;                     // [4][128] f32
constexpr int OFF_SE    = 149952;                     // [80] f32
constexpr int OFF_SSUM  = 150272;                     // [16] f32
constexpr int OFF_WSUM  = 150336;                     // [4][50] f32
constexpr int OFF_WB    = 151136;                     // [4][128] f32
constexpr int OFF_BIAS  = 153184;                     // [1024] f32
constexpr int LDS_BYTES = 157280;                     // ~154 KB -> 2 WGs / WGP

// ---------------- helpers ----------------
struct alignas(16) U4 { unsigned x, y, z, w; };

// Native VALU f32<->bf16 conversions (RNE) — let ISel pick the hw cvt ops.
static __device__ __forceinline__ __bf16 f2bf(float f) { return (__bf16)f; }
static __device__ __forceinline__ float  bf2f(__bf16 b) { return (float)b; }

// ---- TDM: 1-D contiguous global -> LDS DMA (nbytes multiple of 8) ----
static __device__ __forceinline__ void tdm_copy_1d(unsigned ldsOffBytes, const void* gsrc,
                                                   unsigned nbytes) {
#if __has_builtin(__builtin_amdgcn_tensor_load_to_lds)
  unsigned n8 = nbytes >> 3;                         // 8-byte elements
  unsigned long long ga = (unsigned long long)(size_t)gsrc;
  u32x4 g0;
  g0.x = 1u;                                         // count=1 (valid descriptor)
  g0.y = ldsOffBytes;                                // lds_addr [63:32]
  g0.z = (unsigned)ga;                               // global_addr[31:0]
  g0.w = (unsigned)((ga >> 32) & 0x01FFFFFFu) | 0x80000000u;  // addr[56:32] | type=2
  i32x8 g1;
  g1[0] = (int)(3u << 16);                           // workgroup_mask=0, data_size=3 (8B)
  g1[1] = (int)((n8 & 0xFFFFu) << 16);               // tensor_dim0[15:0] @ bits 63:48
  g1[2] = (int)(((n8 >> 16) & 0xFFFFu) | (1u << 16)); // tensor_dim0[31:16], tensor_dim1=1
  g1[3] = (int)((n8 & 0xFFFFu) << 16);               // tile_dim0 @ bits 127:112
  g1[4] = 1;                                         // tile_dim1=1, tile_dim2=0
  g1[5] = (int)n8;                                   // tensor_dim0_stride low
  g1[6] = 0;
  g1[7] = 0;
  i32x4 z4 = {0, 0, 0, 0};
#if __clang_major__ >= 23
  i32x8 z8 = {0, 0, 0, 0, 0, 0, 0, 0};
  __builtin_amdgcn_tensor_load_to_lds(g0, g1, z4, z4, z8, 0);
#else
  __builtin_amdgcn_tensor_load_to_lds(g0, g1, z4, z4, 0);
#endif
#else
  // Fallback: wave0 cooperative copy
  extern __shared__ char smem_[];
  const U4* s = (const U4*)gsrc;
  U4* d = (U4*)(smem_ + ldsOffBytes);
  for (unsigned i = threadIdx.x & 31; i < (nbytes >> 4); i += 32) d[i] = s[i];
#endif
}
static __device__ __forceinline__ void tdm_wait1() {
#if __has_builtin(__builtin_amdgcn_s_wait_tensorcnt)
  if (threadIdx.x < 32) __builtin_amdgcn_s_wait_tensorcnt((short)1);
#endif
  __syncthreads();
}
static __device__ __forceinline__ void tdm_wait0() {
#if __has_builtin(__builtin_amdgcn_s_wait_tensorcnt)
  if (threadIdx.x < 32) __builtin_amdgcn_s_wait_tensorcnt((short)0);
#endif
  __syncthreads();
}

// A fragment: 16x32 bf16 rows=tokens; act is [token][stride] bf16 in LDS.
static __device__ __forceinline__ v16bf loadA(const __bf16* act, int m0, int kc, int stride) {
  int lane = threadIdx.x & 31;
  int m = lane & 15, hi = lane >> 4;
  const __bf16* base = act + (m0 + m) * stride + kc * 32 + hi * 8;
  union { v16bf v; U4 q[2]; } u;
  u.q[0] = *(const U4*)(base);       // K = kc*32 + hi*8 .. +7   -> VGPR0-3
  u.q[1] = *(const U4*)(base + 16);  // K = kc*32 + 16 + hi*8 .. -> VGPR4-7
  return u.v;
}
// B fragment: 32x16 (KxN); w is transposed weights [n][kpad] bf16 in LDS.
static __device__ __forceinline__ v16bf loadB(const __bf16* w, int nt, int kc, int kpad) {
  int lane = threadIdx.x & 31;
  int n = lane & 15, hi = lane >> 4;
  const __bf16* base = w + (nt * 16 + n) * kpad + kc * 32 + hi * 8;
  union { v16bf v; U4 q[2]; } u;
  u.q[0] = *(const U4*)(base);
  u.q[1] = *(const U4*)(base + 16);
  return u.v;
}

// Per-wave GEMM over all n-tiles for this wave's M-tile, epilogue fused.
static __device__ __forceinline__ void gemmWave(const __bf16* src, int sstride, int kchunks,
                                                const __bf16* wbuf,
                                                __bf16* dst, int dstride, int ntiles,
                                                const float* biasS,      // LDS, padded width
                                                int nout, bool relu,
                                                const float* extraB,     // [batch][128] or null
                                                float* outG, int tok0) { // global f32 mode
  int lane = threadIdx.x & 31;
  int mtile = threadIdx.x >> 5;
  int n = lane & 15, hi = lane >> 4;
  int kpad = kchunks * 32;
  for (int nt = 0; nt < ntiles; ++nt) {
    v8f c = {0.f, 0.f, 0.f, 0.f, 0.f, 0.f, 0.f, 0.f};
    for (int kc = 0; kc < kchunks; ++kc) {
      v16bf a = loadA(src, mtile * 16, kc, sstride);
      v16bf b = loadB(wbuf, nt, kc, kpad);
      c = __builtin_amdgcn_wmma_f32_16x16x32_bf16(false, a, false, b, (short)0, c, false, false);
    }
    int col = nt * 16 + n;
    float bias = biasS[col];                       // zero-padded -> no guard
#pragma unroll
    for (int r = 0; r < 8; ++r) {
      int m = r + 8 * hi;
      int t = mtile * 16 + m;
      float v = c[r] + bias;
      if (extraB) v += extraB[(t / SEQ) * 128 + col];
      if (relu) v = v > 0.0f ? v : 0.0f;
      if (outG) {
        if (col < nout) outG[(size_t)(tok0 + t) * nout + col] = v;
      } else {
        dst[t * dstride + col] = f2bf(v);
      }
    }
  }
}

// ---------------- prep kernels ----------------
// Transpose+convert all weights once: wimg[l] is [npad][kpad] bf16, [n][k] layout.
__global__ __launch_bounds__(256)
void prep_w(const float* m1_w0, const float* m1_w1, const float* m2_w0, const float* m2_w1,
            const float* at_w0, const float* at_w1, const float* at_w2,
            const float* hp_w0, const float* hp_w1, const float* hp_w2, __bf16* wimg) {
  const int off[11] = {0, 5120, 25600, 41984, 50176, 66560, 82944, 84992, 93184, 109568, 111616};
  const int krows[10] = {13, 150, 100, 100, 100, 100, 100, 50, 100, 100};
  const int fo[10]    = {150, 100, 100, 50, 100, 100, 1, 100, 100, 7};
  const int kpad[10]  = {32, 160, 128, 128, 128, 128, 128, 64, 128, 128};
  const float* Wp[10] = {m1_w0, m1_w1, m2_w0, m2_w1, at_w0, at_w1, at_w2, hp_w0, hp_w1, hp_w2};
  int idx = blockIdx.x * 256 + threadIdx.x;
  if (idx >= WIMG_TOTAL) return;
  int l = 0;
  while (idx >= off[l + 1]) ++l;
  int local = idx - off[l];
  int kp = kpad[l];
  int n = local / kp, k = local - n * kp;
  float v = (n < fo[l] && k < krows[l]) ? Wp[l][(size_t)k * fo[l] + n] : 0.0f;
  wimg[idx] = f2bf(v);
}

// Pre-convert state: x -> bf16 [token][16] (cols 13..15 zero), mask -> f32.
__global__ __launch_bounds__(256)
void prep_x(const float* state, __bf16* xbf, float* maskF) {
  int t = blockIdx.x * 256 + threadIdx.x;
  if (t >= 32768 * 20) return;
  const float* s = state + (size_t)t * 14;
  union { __bf16 h[16]; U4 q[2]; } r;
#pragma unroll
  for (int c = 0; c < 13; ++c) r.h[c] = f2bf(s[c]);
  r.h[13] = f2bf(0.0f); r.h[14] = f2bf(0.0f); r.h[15] = f2bf(0.0f);
  U4* d = (U4*)(xbf + (size_t)t * 16);
  d[0] = r.q[0]; d[1] = r.q[1];
  maskF[t] = s[13];
}

// ---------------- main kernel ----------------
__global__ __launch_bounds__(NT, 2)
void sp_kernel(const float* m1_b0, const float* m1_b1, const float* m2_b0, const float* m2_b1,
               const float* at_b0, const float* at_b1, const float* at_b2,
               const float* hp_b0, const float* hp_b1, const float* hp_b2,
               const float* at_w0, const float* hp_w0,
               const __bf16* wimg, const __bf16* xbf, const float* maskF, float* outG) {
  extern __shared__ char smem[];
  __bf16* wb0   = (__bf16*)(smem + OFF_WBUF0);
  __bf16* wb1   = (__bf16*)(smem + OFF_WBUF1);
  __bf16* actX  = (__bf16*)(smem + OFF_X);
  __bf16* actT1 = (__bf16*)(smem + OFF_T1);
  __bf16* actH1 = (__bf16*)(smem + OFF_H1);
  __bf16* actH2 = (__bf16*)(smem + OFF_H2);
  __bf16* actSC = (__bf16*)(smem + OFF_SC);
  float*  maskS = (float*)(smem + OFF_MASK);
  float*  invMs = (float*)(smem + OFF_MSUM);
  float*  gS    = (float*)(smem + OFF_G);
  float*  gbS   = (float*)(smem + OFF_GB);
  float*  seS   = (float*)(smem + OFF_SE);
  float*  invSs = (float*)(smem + OFF_SSUM);
  float*  wsumS = (float*)(smem + OFF_WSUM);
  float*  wbS   = (float*)(smem + OFF_WB);
  float*  biasS = (float*)(smem + OFF_BIAS);

  const int tid = threadIdx.x;
  const int tok0 = blockIdx.x * TOK;

  // ---- prologue: kick off TDM for first two weight images ----
  if (tid < 32) {
    tdm_copy_1d(OFF_WBUF0, wimg + IMG_OFF[0], IMG_SZ[0] * 2);
    tdm_copy_1d(OFF_WBUF1, wimg + IMG_OFF[1], IMG_SZ[1] * 2);
  }
  // stage x [80][32] (upper 16 cols zero), mask, zero-padded biases
  for (int idx = tid; idx < TOK * 4; idx += NT) {
    int t = idx >> 2, part = idx & 3;
    U4 v = {0u, 0u, 0u, 0u};
    if (part < 2) v = *(const U4*)(xbf + (size_t)(tok0 + t) * 16 + part * 8);
    *(U4*)(actX + t * 32 + part * 8) = v;
  }
  for (int t = tid; t < TOK; t += NT) maskS[t] = maskF[tok0 + t];
  {
    const float* bp[10] = {m1_b0, m1_b1, m2_b0, m2_b1, at_b0, at_b1, at_b2, hp_b0, hp_b1, hp_b2};
    for (int idx = tid; idx < BO[10]; idx += NT) {
      int l = 0;
      while (idx >= BO[l + 1]) ++l;
      int c = idx - BO[l];
      biasS[idx] = (c < BFO[l]) ? bp[l][c] : 0.0f;
    }
  }

  // ---- L0: x(pad32) -> t1(150, relu) [buf0] ----
  tdm_wait1();
  gemmWave(actX, 32, 1, wb0, actT1, 160, 10, biasS + BO[0], 150, true, nullptr, nullptr, tok0);
  __syncthreads();
  if (tid < 32) tdm_copy_1d(OFF_WBUF0, wimg + IMG_OFF[2], IMG_SZ[2] * 2);

  // ---- L1: t1 -> h1(100, relu) [buf1] ----
  tdm_wait1();
  gemmWave(actT1, 160, 5, wb1, actH1, 128, 8, biasS + BO[1], 100, true, nullptr, nullptr, tok0);
  __syncthreads();
  if (tid < 32) tdm_copy_1d(OFF_WBUF1, wimg + IMG_OFF[3], IMG_SZ[3] * 2);

  // ---- L2: h1 -> h2a(100, relu) [buf0] ----
  tdm_wait1();
  gemmWave(actH1, 128, 4, wb0, actT1, 128, 8, biasS + BO[2], 100, true, nullptr, nullptr, tok0);
  __syncthreads();
  if (tid < 32) tdm_copy_1d(OFF_WBUF0, wimg + IMG_OFF[4], IMG_SZ[4] * 2);

  // ---- L3: h2a -> h2(50) [buf1] ----
  tdm_wait1();
  gemmWave(actT1, 128, 4, wb1, actH2, 64, 4, biasS + BO[3], 50, false, nullptr, nullptr, tok0);
  __syncthreads();
  if (tid < 32) tdm_copy_1d(OFF_WBUF1, wimg + IMG_OFF[5], IMG_SZ[5] * 2);

  // ---- reductions: g[b][f], then gb = g @ at_w0[100:200] ----
  if (tid < NB) {
    float s = 0.0f;
    for (int n = 0; n < SEQ; ++n) s += maskS[tid * SEQ + n];
    invMs[tid] = (s > 0.0f) ? 1.0f / s : 0.0f;
  }
  __syncthreads();
  for (int idx = tid; idx < NB * 100; idx += NT) {
    int b = idx / 100, f = idx - b * 100;
    float acc = 0.0f;
    for (int n = 0; n < SEQ; ++n) {
      int t = b * SEQ + n;
      acc += maskS[t] * bf2f(actH1[t * 128 + f]);
    }
    gS[idx] = acc * invMs[b];
  }
  __syncthreads();
  for (int idx = tid; idx < NB * 128; idx += NT) {
    int b = idx >> 7, j = idx & 127;
    float acc = 0.0f;
    if (j < 100)
      for (int f = 0; f < 100; ++f) acc += gS[b * 100 + f] * at_w0[(size_t)(100 + f) * 100 + j];
    gbS[idx] = acc;
  }
  __syncthreads();

  // ---- L4: h1 @ at_w0[:100] + gb -> a1(100, relu) [buf0] ----
  tdm_wait1();
  gemmWave(actH1, 128, 4, wb0, actT1, 128, 8, biasS + BO[4], 100, true, gbS, nullptr, tok0);
  __syncthreads();
  if (tid < 32) tdm_copy_1d(OFF_WBUF0, wimg + IMG_OFF[6], IMG_SZ[6] * 2);

  // ---- L5: a1 -> a2(100, relu) [buf1] ----
  tdm_wait1();
  gemmWave(actT1, 128, 4, wb1, actH1, 128, 8, biasS + BO[5], 100, true, nullptr, nullptr, tok0);
  __syncthreads();
  if (tid < 32) tdm_copy_1d(OFF_WBUF1, wimg + IMG_OFF[7], IMG_SZ[7] * 2);

  // ---- L6: a2 -> score(1) [buf0] ----
  tdm_wait1();
  gemmWave(actH1, 128, 4, wb0, actSC, 16, 1, biasS + BO[6], 1, false, nullptr, nullptr, tok0);
  __syncthreads();
  if (tid < 32) tdm_copy_1d(OFF_WBUF0, wimg + IMG_OFF[8], IMG_SZ[8] * 2);

  // ---- masked softmax, wsum, wb = wsum @ hp_w0[50:100] ----
  for (int t = tid; t < TOK; t += NT) {
    float sc = bf2f(actSC[t * 16]) * maskS[t];
    seS[t] = (sc != 0.0f) ? __expf(sc) : 0.0f;
  }
  __syncthreads();
  if (tid < NB) {
    float s = 0.0f;
    for (int n = 0; n < SEQ; ++n) s += seS[tid * SEQ + n];
    invSs[tid] = (s > 0.0f) ? 1.0f / s : 0.0f;
  }
  __syncthreads();
  for (int t = tid; t < TOK; t += NT) seS[t] = seS[t] * invSs[t / SEQ];
  __syncthreads();
  for (int idx = tid; idx < NB * 50; idx += NT) {
    int b = idx / 50, f = idx - b * 50;
    float acc = 0.0f;
    for (int n = 0; n < SEQ; ++n) {
      int t = b * SEQ + n;
      acc += seS[t] * bf2f(actH2[t * 64 + f]);
    }
    wsumS[idx] = acc;
  }
  __syncthreads();
  for (int idx = tid; idx < NB * 128; idx += NT) {
    int b = idx >> 7, j = idx & 127;
    float acc = 0.0f;
    if (j < 100)
      for (int f = 0; f < 50; ++f) acc += wsumS[b * 50 + f] * hp_w0[(size_t)(50 + f) * 100 + j];
    wbS[idx] = acc;
  }
  __syncthreads();

  // ---- L7: h2 @ hp_w0[:50] + wb -> p1(100, relu) [buf1] ----
  tdm_wait1();
  gemmWave(actH2, 64, 2, wb1, actT1, 128, 8, biasS + BO[7], 100, true, wbS, nullptr, tok0);
  __syncthreads();
  if (tid < 32) tdm_copy_1d(OFF_WBUF1, wimg + IMG_OFF[9], IMG_SZ[9] * 2);

  // ---- L8: p1 -> p2(100, relu) [buf0] ----
  tdm_wait1();
  gemmWave(actT1, 128, 4, wb0, actH1, 128, 8, biasS + BO[8], 100, true, nullptr, nullptr, tok0);
  __syncthreads();

  // ---- L9: p2 -> out(7) f32 global [buf1] ----
  tdm_wait0();
  gemmWave(actH1, 128, 4, wb1, nullptr, 0, 1, biasS + BO[9], 7, false, nullptr, outG, tok0);
}

// ---------------- host launch ----------------
extern "C" void kernel_launch(void* const* d_in, const int* in_sizes, int n_in,
                              void* d_out, int out_size, void* d_ws, size_t ws_size,
                              hipStream_t stream) {
  const float* m1_w0 = (const float*)d_in[0];
  const float* m1_b0 = (const float*)d_in[1];
  const float* m1_w1 = (const float*)d_in[2];
  const float* m1_b1 = (const float*)d_in[3];
  const float* m2_w0 = (const float*)d_in[4];
  const float* m2_b0 = (const float*)d_in[5];
  const float* m2_w1 = (const float*)d_in[6];
  const float* m2_b1 = (const float*)d_in[7];
  const float* at_w0 = (const float*)d_in[8];
  const float* at_b0 = (const float*)d_in[9];
  const float* at_w1 = (const float*)d_in[10];
  const float* at_b1 = (const float*)d_in[11];
  const float* at_w2 = (const float*)d_in[12];
  const float* at_b2 = (const float*)d_in[13];
  const float* hp_w0 = (const float*)d_in[14];
  const float* hp_b0 = (const float*)d_in[15];
  const float* hp_w1 = (const float*)d_in[16];
  const float* hp_b1 = (const float*)d_in[17];
  const float* hp_w2 = (const float*)d_in[18];
  const float* hp_b2 = (const float*)d_in[19];
  const float* state = (const float*)d_in[20];
  float* out = (float*)d_out;

  __bf16* wimg = (__bf16*)((char*)d_ws + WS_WIMG);
  __bf16* xbf  = (__bf16*)((char*)d_ws + WS_XBF);
  float*  maskF = (float*)((char*)d_ws + WS_MASK);

  prep_w<<<(WIMG_TOTAL + 255) / 256, 256, 0, stream>>>(
      m1_w0, m1_w1, m2_w0, m2_w1, at_w0, at_w1, at_w2, hp_w0, hp_w1, hp_w2, wimg);
  prep_x<<<(32768 * 20 + 255) / 256, 256, 0, stream>>>(state, xbf, maskF);

  (void)hipFuncSetAttribute((const void*)sp_kernel,
                            hipFuncAttributeMaxDynamicSharedMemorySize, LDS_BYTES);
  dim3 grid((32768 * 20) / TOK), block(NT);
  sp_kernel<<<grid, block, LDS_BYTES, stream>>>(
      m1_b0, m1_b1, m2_b0, m2_b1, at_b0, at_b1, at_b2, hp_b0, hp_b1, hp_b2,
      at_w0, hp_w0, wimg, xbf, maskF, out);
}
